// VisnetBlock_2259152798302
// MI455X (gfx1250) — matrix operand order repaired and checked by reading, not compiled, not run on hardware
//
#include <hip/hip_runtime.h>
#include <math.h>

// ---------------- problem constants (from reference) ----------------
#define Nn 4096
#define Ee 65536
#define Cc 128
#define Hh 8
#define Ll 4
#define Rr 32
#define VD 8
#define CUTR 5.0f
#define CH 8192        // edge chunk for the vector-rejection GEMMs

typedef __attribute__((ext_vector_type(16))) __bf16 v16bf;
typedef __attribute__((ext_vector_type(2)))  __bf16 v2bf;
typedef __attribute__((ext_vector_type(2)))  float  v2f;
typedef __attribute__((ext_vector_type(8)))  float  v8f;
typedef unsigned int uint32;

// f32x2 -> packed bf16x2 (RNE); lowers to v_cvt_pk_bf16_f32 on gfx1250
__device__ __forceinline__ uint32 pack2bf(float lo, float hi) {
  v2f f; f.x = lo; f.y = hi;
  v2bf b = __builtin_convertvector(f, v2bf);
  union { v2bf b; uint32 u; } x; x.b = b;
  return x.u;
}
__device__ __forceinline__ float siluf(float x) { return x / (1.0f + __expf(-x)); }

union FragU { uint4 q[2]; v16bf bf; };

// =====================================================================
// WMMA GEMM:  C[M,NCOL] = epi( A[M,KD] @ B[KD,NCOL] + bias )
// NCOL, KD compile-time -> all row offsets fold into load immediates.
// A rows optionally gathered:  physical_row = gather[m>>3]*8 + (m&7)
// Requirements: M%64==0, NCOL%64==0, KD%32==0. 128 threads (4 waves),
// 64x64 block tile, each wave owns 32x32 (4 WMMA accumulators).
// LDS: packed bf16 pairs; A row-major (stride 20 u32), B col-major.
// Fragments: 2 x ds_load_b128 each. Next tile prefetched to registers
// before the WMMAs of the current one.
// EPI: 0 = none, 1 = +bias, 2 = silu(+bias)
// =====================================================================
template <int EPI, bool GATHER, int NCOL, int KD>
__global__ __launch_bounds__(128)
void k_gemm(const float* __restrict__ A, const float* __restrict__ B,
            const float* __restrict__ bias, float* __restrict__ Cmat,
            int M, const int* __restrict__ gather)
{
  __shared__ uint32 As[64 * 20];   // 64 rows x 16 pairs (+4 pad)
  __shared__ uint32 Bs[64 * 20];   // 64 cols x 16 pairs (+4 pad)

  const int tid  = threadIdx.x;
  const int lane = tid & 31;
  const int wave = tid >> 5;
  const int wm   = (wave >> 1) << 5;     // 0 / 32
  const int wn   = (wave & 1) << 5;      // 0 / 32
  const int bm   = blockIdx.y * 64;
  const int bn   = blockIdx.x * 64;
  const int half = lane >> 4;            // 0/1
  const int mr   = lane & 15;

  // ---- A staging: 4 float4 per thread; column fixed, rows i*16+(tid>>3) ----
  const int acol = (tid * 4) & 31;       // multiple of 4
  const float* ap[4];
#pragma unroll
  for (int i = 0; i < 4; ++i) {
    int r = bm + i * 16 + (tid >> 3);
    int grow = GATHER ? (gather[r >> 3] * VD + (r & 7)) : r;
    ap[i] = A + (size_t)grow * KD + acol;
  }
  // ---- B staging: one column per thread, 16 k-values ----
  const int kg = (tid >> 6) << 4;        // 0 or 16
  const float* bp = B + (size_t)kg * NCOL + bn + (tid & 63);

  float4 fa[4];
  float  fb[16];
  const int KT = KD >> 5;

  auto loadA = [&](int kt) {
#pragma unroll
    for (int i = 0; i < 4; ++i)
      fa[i] = *(const float4*)&ap[i][kt * 32];
  };
  auto loadB = [&](int kt) {
#pragma unroll
    for (int j = 0; j < 16; ++j)
      fb[j] = bp[(size_t)kt * 32 * NCOL + j * NCOL];   // j*NCOL -> immediate
  };

  v8f c00 = {}, c01 = {}, c10 = {}, c11 = {};

  loadA(0);
  loadB(0);

  for (int kt = 0; kt < KT; ++kt) {
    // ---- store staged tile to LDS (packed bf16 pairs) ----
#pragma unroll
    for (int i = 0; i < 4; ++i) {
      int row = i * 16 + (tid >> 3);
      uint2 u;
      u.x = pack2bf(fa[i].x, fa[i].y);
      u.y = pack2bf(fa[i].z, fa[i].w);
      *(uint2*)&As[row * 20 + (acol >> 1)] = u;
    }
    {
      uint4 p0, p1;
      p0.x = pack2bf(fb[0], fb[1]);   p0.y = pack2bf(fb[2], fb[3]);
      p0.z = pack2bf(fb[4], fb[5]);   p0.w = pack2bf(fb[6], fb[7]);
      p1.x = pack2bf(fb[8], fb[9]);   p1.y = pack2bf(fb[10], fb[11]);
      p1.z = pack2bf(fb[12], fb[13]); p1.w = pack2bf(fb[14], fb[15]);
      *(uint4*)&Bs[(tid & 63) * 20 + (kg >> 1)]     = p0;
      *(uint4*)&Bs[(tid & 63) * 20 + (kg >> 1) + 4] = p1;
    }
    __syncthreads();

    // ---- prefetch next tile (hides global latency under WMMAs) ----
    if (kt + 1 < KT) { loadA(kt + 1); loadB(kt + 1); }

    // ---- fragments: two b128 LDS loads each ----
    FragU a0, a1, b0, b1;
    a0.q[0] = *(const uint4*)&As[(wm + mr) * 20 + half * 4];
    a0.q[1] = *(const uint4*)&As[(wm + mr) * 20 + 8 + half * 4];
    a1.q[0] = *(const uint4*)&As[(wm + 16 + mr) * 20 + half * 4];
    a1.q[1] = *(const uint4*)&As[(wm + 16 + mr) * 20 + 8 + half * 4];
    b0.q[0] = *(const uint4*)&Bs[(wn + mr) * 20 + half * 4];
    b0.q[1] = *(const uint4*)&Bs[(wn + mr) * 20 + 8 + half * 4];
    b1.q[0] = *(const uint4*)&Bs[(wn + 16 + mr) * 20 + half * 4];
    b1.q[1] = *(const uint4*)&Bs[(wn + 16 + mr) * 20 + 8 + half * 4];

    c00 = __builtin_amdgcn_wmma_f32_16x16x32_bf16(false, a0.bf, false, b0.bf, (short)0, c00, false, false);
    c01 = __builtin_amdgcn_wmma_f32_16x16x32_bf16(false, a0.bf, false, b1.bf, (short)0, c01, false, false);
    c10 = __builtin_amdgcn_wmma_f32_16x16x32_bf16(false, a1.bf, false, b0.bf, (short)0, c10, false, false);
    c11 = __builtin_amdgcn_wmma_f32_16x16x32_bf16(false, a1.bf, false, b1.bf, (short)0, c11, false, false);
    __syncthreads();
  }

  // ---- epilogue: D layout VGPR r -> (M = r + half*8, N = mr) ----
  v8f accs[4] = { c00, c01, c10, c11 };
#pragma unroll
  for (int q = 0; q < 4; ++q) {
    int fm = q >> 1, fn = q & 1;
    int col = bn + wn + fn * 16 + mr;
    float bv = (EPI >= 1) ? bias[col] : 0.0f;
#pragma unroll
    for (int r = 0; r < 8; ++r) {
      int row = bm + wm + fm * 16 + half * 8 + r;
      float vv = accs[q][r] + bv;
      if (EPI == 2) vv = siluf(vv);
      Cmat[(size_t)row * NCOL + col] = vv;
    }
  }
}

template <int EPI, bool GATHER, int NCOL, int KD>
static inline void launch_gemm(const float* A, const float* B, const float* bias, float* Cm,
                               int M, const int* gather, hipStream_t s)
{
  dim3 g((unsigned)(NCOL / 64), (unsigned)(M / 64)), b(128);
  k_gemm<EPI, GATHER, NCOL, KD><<<g, b, 0, s>>>(A, B, bias, Cm, M, gather);
}

// =====================================================================
// elementwise / reduction kernels
// =====================================================================
__global__ void k_edge_geom(const float* __restrict__ ev, float* dist, float* cutv,
                            float* dij, float* rbf)
{
  int e = blockIdx.x * blockDim.x + threadIdx.x;
  if (e >= Ee) return;
  float x = ev[3*e], y = ev[3*e+1], z = ev[3*e+2];
  float d = sqrtf(x*x + y*y + z*z + 1e-16f);
  float co = (d < CUTR) ? 0.5f * (cosf(3.14159265358979f * d / CUTR) + 1.0f) : 0.0f;
  dist[e] = d; cutv[e] = co;
  float inv = 1.0f / (d + 1e-8f);
  float ux = x*inv, uy = y*inv, uz = z*inv;
  const float s3 = 1.7320508075688772f;
  float* dd = dij + (size_t)e * VD;
  dd[0]=ux; dd[1]=uy; dd[2]=uz; dd[3]=s3*ux*uy; dd[4]=s3*ux*uz;
  dd[5]=uz*uz - 0.5f*ux*ux - 0.5f*uy*uy; dd[6]=s3*uy*uz;
  dd[7]=0.5f*s3*(ux*ux - uy*uy);
  float ecut = expf(-CUTR);
  float tt = (2.0f / Rr) * (1.0f - ecut);
  float beta = 1.0f / (tt * tt);
  float ex = __expf(-(5.0f / CUTR) * d);
  float* rr = rbf + (size_t)e * Rr;
  for (int r = 0; r < Rr; ++r) {
    float mean = ecut + (1.0f - ecut) * ((float)r / (Rr - 1));
    float dm = ex - mean;
    rr[r] = co * __expf(-beta * dm * dm);
  }
}

__global__ void k_gather_embed(const float* __restrict__ emb, const int* __restrict__ spec,
                               float* __restrict__ x)
{
  int i = blockIdx.x * blockDim.x + threadIdx.x;
  if (i >= Nn * Cc) return;
  int n = i >> 7, c = i & 127;
  x[i] = emb[spec[n] * Cc + c];
}

__global__ void k_ne_agg(const float* __restrict__ ne_emb, const int* __restrict__ spec,
                         const int* __restrict__ snd, const int* __restrict__ rcv,
                         const float* __restrict__ Wb, const float* __restrict__ cutv,
                         float* __restrict__ agg)
{
  int i = blockIdx.x * blockDim.x + threadIdx.x;
  if (i >= Ee * Cc) return;
  int e = i >> 7, c = i & 127;
  float m = ne_emb[spec[snd[e]] * Cc + c] * Wb[i] * cutv[e];
  atomicAdd(&agg[rcv[e] * Cc + c], m);
}

__global__ void k_concat(const float* __restrict__ x, const float* __restrict__ agg,
                         float* __restrict__ xcat)
{
  int i = blockIdx.x * blockDim.x + threadIdx.x;
  if (i >= Nn * 2 * Cc) return;
  int n = i >> 8, c = i & 255;
  xcat[i] = (c < Cc) ? x[n * Cc + c] : agg[n * Cc + (c - Cc)];
}

__global__ void k_f_init(const float* __restrict__ x, const int* __restrict__ snd,
                         const int* __restrict__ rcv, const float* __restrict__ t,
                         float* __restrict__ f)
{
  int i = blockIdx.x * blockDim.x + threadIdx.x;
  if (i >= Ee * Cc) return;
  int e = i >> 7, c = i & 127;
  f[i] = (x[snd[e] * Cc + c] + x[rcv[e] * Cc + c]) * t[i];
}

__global__ __launch_bounds__(128)
void k_layernorm(const float* __restrict__ x, const float* __restrict__ g,
                 const float* __restrict__ b, float* __restrict__ y)
{
  __shared__ float red[128];
  int n = blockIdx.x, c = threadIdx.x;
  float v = x[(size_t)n * Cc + c];
  red[c] = v; __syncthreads();
  for (int s = 64; s > 0; s >>= 1) { if (c < s) red[c] += red[c + s]; __syncthreads(); }
  float mu = red[0] * (1.0f / Cc); __syncthreads();
  float dv = v - mu;
  red[c] = dv * dv; __syncthreads();
  for (int s = 64; s > 0; s >>= 1) { if (c < s) red[c] += red[c + s]; __syncthreads(); }
  float var = red[0] * (1.0f / Cc);
  y[(size_t)n * Cc + c] = dv * rsqrtf(var + 1e-5f) * g[c] + b[c];
}

__global__ void k_vecln(const float* __restrict__ vec, const float* __restrict__ w,
                        float* __restrict__ vecl)
{
  int i = blockIdx.x * blockDim.x + threadIdx.x;
  if (i >= Nn * VD * Cc) return;
  vecl[i] = vec[i] * w[i & 127];
}

__global__ void k_vecdot(const float* __restrict__ v123, float* __restrict__ vdot)
{
  int i = blockIdx.x * blockDim.x + threadIdx.x;
  if (i >= Nn * Cc) return;
  int n = i >> 7, c = i & 127;
  float s = 0.0f;
  for (int d = 0; d < VD; ++d) {
    size_t base = ((size_t)n * VD + d) * (3 * Cc);
    s += v123[base + c] * v123[base + Cc + c];
  }
  vdot[i] = s;
}

__global__ void k_attn(const float* __restrict__ q, const float* __restrict__ kk,
                       const float* __restrict__ dk, const int* __restrict__ snd,
                       const int* __restrict__ rcv, const float* __restrict__ cutv,
                       float* __restrict__ attn)
{
  int t = blockIdx.x * blockDim.x + threadIdx.x;
  if (t >= Ee * Hh) return;
  int e = t >> 3, h = t & 7;
  int qb = rcv[e] * Cc + h * 16, kb = snd[e] * Cc + h * 16, db = e * Cc + h * 16;
  float s = 0.0f;
  for (int i = 0; i < 16; ++i) s += q[qb + i] * kk[kb + i] * dk[db + i];
  attn[t] = siluf(s) * cutv[e];
}

__global__ void k_vj(const float* __restrict__ v, const float* __restrict__ dv,
                     const float* __restrict__ attn, const int* __restrict__ snd,
                     float* __restrict__ vj)
{
  int i = blockIdx.x * blockDim.x + threadIdx.x;
  if (i >= Ee * Cc) return;
  int e = i >> 7, c = i & 127;
  vj[i] = v[snd[e] * Cc + c] * dv[i] * attn[e * Hh + (c >> 4)];
}

__global__ void k_msg_agg(const float* __restrict__ vj, const float* __restrict__ s12,
                          const float* __restrict__ vecl, const float* __restrict__ dij,
                          const int* __restrict__ snd, const int* __restrict__ rcv,
                          float* __restrict__ xagg, float* __restrict__ vagg)
{
  int i = blockIdx.x * blockDim.x + threadIdx.x;
  if (i >= Ee * Cc) return;
  int e = i >> 7, c = i & 127;
  int rv = rcv[e], sn = snd[e];
  atomicAdd(&xagg[rv * Cc + c], vj[i]);
  float s1 = s12[(size_t)e * 2 * Cc + c];
  float s2 = s12[(size_t)e * 2 * Cc + Cc + c];
  for (int d = 0; d < VD; ++d) {
    float vm = vecl[((size_t)sn * VD + d) * Cc + c] * s1 + s2 * dij[(size_t)e * VD + d];
    atomicAdd(&vagg[((size_t)rv * VD + d) * Cc + c], vm);
  }
}

__global__ void k_update(const float* __restrict__ vdot, const float* __restrict__ o123,
                         const float* __restrict__ v123, const float* __restrict__ vagg,
                         float* __restrict__ x, float* __restrict__ vec)
{
  int i = blockIdx.x * blockDim.x + threadIdx.x;
  if (i >= Nn * Cc) return;
  int n = i >> 7, c = i & 127;
  float o1 = o123[(size_t)n * 3 * Cc + c];
  float o2 = o123[(size_t)n * 3 * Cc + Cc + c];
  float o3 = o123[(size_t)n * 3 * Cc + 2 * Cc + c];
  x[i] += vdot[i] * o2 + o3;
  for (int d = 0; d < VD; ++d) {
    size_t vi = ((size_t)n * VD + d) * Cc + c;
    float v3 = v123[((size_t)n * VD + d) * (3 * Cc) + 2 * Cc + c];
    vec[vi] += v3 * o1 + vagg[vi];
  }
}

__global__ void k_rej_wdot(int e0, const float* __restrict__ w1r, const float* __restrict__ w2r,
                           const float* __restrict__ dij, float* __restrict__ wdot)
{
  int i = blockIdx.x * blockDim.x + threadIdx.x;
  if (i >= CH * Cc) return;
  int el = i >> 7, c = i & 127;
  int e = e0 + el;
  float dj[VD], w1[VD], w2[VD];
  for (int d = 0; d < VD; ++d) dj[d] = dij[(size_t)e * VD + d];
  float p1 = 0.0f, p2 = 0.0f;
  for (int d = 0; d < VD; ++d) {
    w1[d] = w1r[((size_t)el * VD + d) * Cc + c];
    w2[d] = w2r[((size_t)el * VD + d) * Cc + c];
    p1 += w1[d] * dj[d];
    p2 += w2[d] * dj[d];  // rejection against -dij == rejection against dij
  }
  float s = 0.0f;
  for (int d = 0; d < VD; ++d) s += (w1[d] - p1 * dj[d]) * (w2[d] - p2 * dj[d]);
  wdot[(size_t)e * Cc + c] = s;
}

__global__ void k_f_update(const float* __restrict__ ft, const float* __restrict__ wdot,
                           float* __restrict__ f)
{
  int i = blockIdx.x * blockDim.x + threadIdx.x;
  if (i >= Ee * Cc) return;
  f[i] += ft[i] * wdot[i];
}

__global__ void k_outvec(const float* __restrict__ vec, const float* __restrict__ w,
                         float* __restrict__ out)
{
  int i = blockIdx.x * blockDim.x + threadIdx.x;
  if (i >= Nn * VD * Cc) return;
  out[i] = vec[i] * w[i & 127];
}

// =====================================================================
// host driver
// =====================================================================
struct LayerP {
  const float *dk_W,*dk_b,*dv_W,*dv_b,*f_W,*f_b,*k_W,*k_b,*ln_b,*ln_g,
              *o_W,*o_b,*q_W,*q_b,*s_W,*s_b,*v_W,*v_b,*vec_W,*vln_w,*ws_W,*wt_W;
};

extern "C" void kernel_launch(void* const* d_in, const int* in_sizes, int n_in,
                              void* d_out, int out_size, void* d_ws, size_t ws_size,
                              hipStream_t stream)
{
  (void)in_sizes; (void)n_in; (void)out_size; (void)ws_size;
  const float* ev      = (const float*)d_in[0];
  const int*   species = (const int*)d_in[1];
  const int*   snd     = (const int*)d_in[2];
  const int*   rcv     = (const int*)d_in[3];

  // params flattened in JAX pytree (sorted-key) order
  int pi = 4;
  auto nextp = [&]() { return (const float*)d_in[pi++]; };
  const float* ee_W = nextp(); const float* ee_b = nextp(); const float* embed = nextp();
  LayerP lp[Ll];
  for (int l = 0; l < Ll; ++l) {
    bool last = (l == Ll - 1);
    lp[l].dk_W = nextp(); lp[l].dk_b = nextp(); lp[l].dv_W = nextp(); lp[l].dv_b = nextp();
    if (!last) { lp[l].f_W = nextp(); lp[l].f_b = nextp(); } else { lp[l].f_W = nullptr; lp[l].f_b = nullptr; }
    lp[l].k_W = nextp(); lp[l].k_b = nextp(); lp[l].ln_b = nextp(); lp[l].ln_g = nextp();
    lp[l].o_W = nextp(); lp[l].o_b = nextp(); lp[l].q_W = nextp(); lp[l].q_b = nextp();
    lp[l].s_W = nextp(); lp[l].s_b = nextp(); lp[l].v_W = nextp(); lp[l].v_b = nextp();
    lp[l].vec_W = nextp(); lp[l].vln_w = nextp();
    if (!last) { lp[l].ws_W = nextp(); lp[l].wt_W = nextp(); } else { lp[l].ws_W = nullptr; lp[l].wt_W = nullptr; }
  }
  const float* ne_comb_W = nextp(); const float* ne_comb_b = nextp();
  const float* ne_dist_W = nextp(); const float* ne_dist_b = nextp();
  const float* ne_embed  = nextp();
  const float* out_b = nextp(); const float* out_g = nextp(); const float* ovln_w = nextp();

  // ---- workspace carve-up (~450 MB) ----
  char* wsb = (char*)d_ws; size_t off = 0;
  auto alloc = [&](size_t nf) { float* p = (float*)(wsb + off); off += ((nf * 4 + 255) / 256) * 256; return p; };
  float* dist = alloc(Ee);               float* cutv = alloc(Ee);
  float* rbf  = alloc((size_t)Ee * Rr);  float* dij  = alloc((size_t)Ee * VD);
  float* xbuf = alloc((size_t)Nn * Cc);  float* aggb = alloc((size_t)Nn * Cc);
  float* xcat = alloc((size_t)Nn * 2 * Cc);
  float* fbuf = alloc((size_t)Ee * Cc);  float* tmpE = alloc((size_t)Ee * Cc);
  float* xl   = alloc((size_t)Nn * Cc);  float* vecl = alloc((size_t)Nn * VD * Cc);
  float* qb   = alloc((size_t)Nn * Cc);  float* kb = alloc((size_t)Nn * Cc);  float* vb = alloc((size_t)Nn * Cc);
  float* dkb  = alloc((size_t)Ee * Cc);  float* dvb = alloc((size_t)Ee * Cc);
  float* v123 = alloc((size_t)Nn * VD * 3 * Cc);
  float* vdot = alloc((size_t)Nn * Cc);  float* attn = alloc((size_t)Ee * Hh);
  float* vj   = alloc((size_t)Ee * Cc);  float* s12  = alloc((size_t)Ee * 2 * Cc);
  float* xagg = alloc((size_t)Nn * Cc);  float* vagg = alloc((size_t)Nn * VD * Cc);
  float* o123 = alloc((size_t)Nn * 3 * Cc);
  float* wdot = alloc((size_t)Ee * Cc);
  float* w1r  = alloc((size_t)CH * VD * Cc);  float* w2r = alloc((size_t)CH * VD * Cc);
  float* vec  = alloc((size_t)Nn * VD * Cc);

  auto nb = [](long n) { return (unsigned)((n + 255) / 256); };

  hipMemsetAsync(vec,  0, (size_t)Nn * VD * Cc * 4, stream);
  hipMemsetAsync(aggb, 0, (size_t)Nn * Cc * 4, stream);

  // ---- geometry + embedding front-end ----
  k_edge_geom<<<nb(Ee), 256, 0, stream>>>(ev, dist, cutv, dij, rbf);
  k_gather_embed<<<nb(Nn * Cc), 256, 0, stream>>>(embed, species, xbuf);
  launch_gemm<1, false, 128, 32>(rbf, ne_dist_W, ne_dist_b, tmpE, Ee, nullptr, stream);
  k_ne_agg<<<nb((long)Ee * Cc), 256, 0, stream>>>(ne_embed, species, snd, rcv, tmpE, cutv, aggb);
  k_concat<<<nb((long)Nn * 2 * Cc), 256, 0, stream>>>(xbuf, aggb, xcat);
  launch_gemm<1, false, 128, 256>(xcat, ne_comb_W, ne_comb_b, xbuf, Nn, nullptr, stream);
  launch_gemm<1, false, 128, 32>(rbf, ee_W, ee_b, tmpE, Ee, nullptr, stream);
  k_f_init<<<nb((long)Ee * Cc), 256, 0, stream>>>(xbuf, snd, rcv, tmpE, fbuf);

  // ---- message-passing layers ----
  for (int l = 0; l < Ll; ++l) {
    k_layernorm<<<Nn, 128, 0, stream>>>(xbuf, lp[l].ln_g, lp[l].ln_b, xl);
    k_vecln<<<nb((long)Nn * VD * Cc), 256, 0, stream>>>(vec, lp[l].vln_w, vecl);
    launch_gemm<1, false, 128, 128>(xl, lp[l].q_W, lp[l].q_b, qb, Nn, nullptr, stream);
    launch_gemm<1, false, 128, 128>(xl, lp[l].k_W, lp[l].k_b, kb, Nn, nullptr, stream);
    launch_gemm<1, false, 128, 128>(xl, lp[l].v_W, lp[l].v_b, vb, Nn, nullptr, stream);
    launch_gemm<2, false, 128, 128>(fbuf, lp[l].dk_W, lp[l].dk_b, dkb, Ee, nullptr, stream);
    launch_gemm<2, false, 128, 128>(fbuf, lp[l].dv_W, lp[l].dv_b, dvb, Ee, nullptr, stream);
    launch_gemm<0, false, 384, 128>(vecl, lp[l].vec_W, nullptr, v123, Nn * VD, nullptr, stream);
    k_vecdot<<<nb((long)Nn * Cc), 256, 0, stream>>>(v123, vdot);
    k_attn<<<nb((long)Ee * Hh), 256, 0, stream>>>(qb, kb, dkb, snd, rcv, cutv, attn);
    k_vj<<<nb((long)Ee * Cc), 256, 0, stream>>>(vb, dvb, attn, snd, vj);
    launch_gemm<2, false, 256, 128>(vj, lp[l].s_W, lp[l].s_b, s12, Ee, nullptr, stream);
    hipMemsetAsync(xagg, 0, (size_t)Nn * Cc * 4, stream);
    hipMemsetAsync(vagg, 0, (size_t)Nn * VD * Cc * 4, stream);
    k_msg_agg<<<nb((long)Ee * Cc), 256, 0, stream>>>(vj, s12, vecl, dij, snd, rcv, xagg, vagg);
    launch_gemm<1, false, 384, 128>(xagg, lp[l].o_W, lp[l].o_b, o123, Nn, nullptr, stream);
    k_update<<<nb((long)Nn * Cc), 256, 0, stream>>>(vdot, o123, v123, vagg, xbuf, vec);

    if (l < Ll - 1) {
      for (int ch = 0; ch < Ee / CH; ++ch) {
        launch_gemm<0, true, 128, 128>(vecl, lp[l].wt_W, nullptr, w1r, CH * VD, rcv + ch * CH, stream);
        launch_gemm<0, true, 128, 128>(vecl, lp[l].ws_W, nullptr, w2r, CH * VD, snd + ch * CH, stream);
        k_rej_wdot<<<nb((long)CH * Cc), 256, 0, stream>>>(ch * CH, w1r, w2r, dij, wdot);
      }
      launch_gemm<2, false, 128, 128>(fbuf, lp[l].f_W, lp[l].f_b, tmpE, Ee, nullptr, stream);
      k_f_update<<<nb((long)Ee * Cc), 256, 0, stream>>>(tmpE, wdot, fbuf);
    }
  }

  // ---- output heads ----
  k_layernorm<<<Nn, 128, 0, stream>>>(xbuf, out_g, out_b, (float*)d_out);
  k_outvec<<<nb((long)Nn * VD * Cc), 256, 0, stream>>>(vec, ovln_w, (float*)d_out + (size_t)Nn * Cc);
}
